// DifferentialEntropyRegularization_70935679860862
// MI455X (gfx1250) — matrix operand order repaired and checked by reading, not compiled, not run on hardware
//
#include <hip/hip_runtime.h>
#include <math.h>

#define N 8192
#define D 512
#define KTOP 5
#define COLSPLITS 4
#define KPAD 520   // 1040B row stride in LDS: ≡16 (mod 256) bytes -> conflict-free banks

typedef __bf16 bfx4  __attribute__((ext_vector_type(4)));
typedef __bf16 bfx8  __attribute__((ext_vector_type(8)));
typedef __bf16 bfx16 __attribute__((ext_vector_type(16)));
typedef float  v8f   __attribute__((ext_vector_type(8)));

// ---------------------------------------------------------------------------
// gfx1250 async copy: global -> LDS, 16 bytes per lane, tracked by ASYNCcnt.
// ---------------------------------------------------------------------------
__device__ __forceinline__ void async_copy_b128(unsigned ldsOff, const void* g) {
  asm volatile("global_load_async_to_lds_b128 %0, %1, off"
               :: "v"(ldsOff), "v"((unsigned long long)(uintptr_t)g)
               : "memory");
}
__device__ __forceinline__ void wait_async0() {
  asm volatile("s_wait_asynccnt 0x0" ::: "memory");
}

// ---------------------------------------------------------------------------
// Kernel 1: row-normalize x (fp32) and split each normalized value into
// bf16 hi + bf16 lo (x = hi + lo) for bf16x3 WMMA fp32-emulation.
// ---------------------------------------------------------------------------
__global__ __launch_bounds__(256) void normalize_split_kernel(
    const float* __restrict__ x, __bf16* __restrict__ xh, __bf16* __restrict__ xl) {
  const int wave = threadIdx.x >> 5;
  const int lane = threadIdx.x & 31;
  const int row  = blockIdx.x * 8 + wave;
  const float* xr = x + (size_t)row * D;

  float4 v[4];
  float ss = 0.f;
#pragma unroll
  for (int j = 0; j < 4; ++j) {
    v[j] = *(const float4*)(xr + j * 128 + lane * 4);
    ss += v[j].x * v[j].x + v[j].y * v[j].y + v[j].z * v[j].z + v[j].w * v[j].w;
  }
#pragma unroll
  for (int off = 16; off >= 1; off >>= 1) ss += __shfl_xor(ss, off, 32);
  const float inv = 1.0f / sqrtf(ss);

#pragma unroll
  for (int j = 0; j < 4; ++j) {
    float e[4] = {v[j].x, v[j].y, v[j].z, v[j].w};
    bfx4 h, l;
#pragma unroll
    for (int q = 0; q < 4; ++q) {
      float xn = e[q] * inv;
      __bf16 hb = (__bf16)xn;
      h[q] = hb;
      l[q] = (__bf16)(xn - (float)hb);
    }
    const int base = row * D + j * 128 + lane * 4;
    *(bfx4*)(xh + base) = h;
    *(bfx4*)(xl + base) = l;
  }
}

// ---------------------------------------------------------------------------
// Kernel 2: fused Gram + running top-5 (values only).
// Each wave owns 16 rows with full-K A (hi+lo) in registers. B tiles are
// fetched ONCE per workgroup via async global->LDS B128 copies (double
// buffered, overlapped with WMMA), so off-WGP B traffic is 64 x 16MB = 1 GB
// total, far under L2 capability; the per-tile barrier keeps all 8 waves in
// lockstep. dots -> top-5 values only, since ||a-b|| = sqrt(2-2*dot).
// ---------------------------------------------------------------------------
__global__ __launch_bounds__(256) void gram_topk_kernel(
    const __bf16* __restrict__ xh, const __bf16* __restrict__ xl,
    float* __restrict__ topvals /* [N][COLSPLITS][KTOP] */) {
  __shared__ __align__(16) __bf16 Bt[2][2][16][KPAD];  // [buf][hi/lo][col][k], 66.5 KB
  __shared__ float sm[8][16][16][KTOP + 1];            // merge scratch, 48 KB

  const int tid   = threadIdx.x;
  const int wave  = tid >> 5;
  const int lane  = tid & 31;
  const int half  = lane >> 4;   // K-half selector for A/B fragments
  const int lan16 = lane & 15;   // A-row / B-col within tile
  const int m0    = blockIdx.x * 128 + wave * 16;
  const int cbeg  = blockIdx.y * (N / COLSPLITS);
  const int cend  = cbeg + (N / COLSPLITS);

  // ---- A fragments, full K=512 (16 chunks of 32), hi and lo, in registers.
  bfx16 Ah[16], Al[16];
  {
    const __bf16* rH = xh + (size_t)(m0 + lan16) * D;
    const __bf16* rL = xl + (size_t)(m0 + lan16) * D;
#pragma unroll
    for (int c = 0; c < 16; ++c) {
      const int k0 = c * 32 + half * 8;
      bfx8 h1 = *(const bfx8*)(rH + k0);
      bfx8 h2 = *(const bfx8*)(rH + k0 + 16);
      Ah[c] = __builtin_shufflevector(h1, h2, 0,1,2,3,4,5,6,7,8,9,10,11,12,13,14,15);
      bfx8 l1 = *(const bfx8*)(rL + k0);
      bfx8 l2 = *(const bfx8*)(rL + k0 + 16);
      Al[c] = __builtin_shufflevector(l1, l2, 0,1,2,3,4,5,6,7,8,9,10,11,12,13,14,15);
    }
  }

  // ---- async-stage one 16-column B tile (hi+lo = 32 KB) into LDS buffer b.
  // 256 threads x 8 issues x 16B = 32 KB, disjoint coverage.
  auto issue_tile = [&](int c0, int b) {
#pragma unroll
    for (int i = 0; i < 8; ++i) {
      const int region = i >> 2;               // 0 = hi, 1 = lo
      const int idx    = (i & 3) * 4096 + tid * 16;  // byte idx in 16KB region
      const int col    = idx >> 10;            // 0..15
      const int kh     = (idx & 1023) >> 1;    // element offset in row
      const __bf16* g  = (region ? xl : xh) + (size_t)(c0 + col) * D + kh;
      const unsigned ldsOff = (unsigned)(uintptr_t)&Bt[b][region][col][kh];
      async_copy_b128(ldsOff, g);
    }
  };

  // ---- per-lane top-5 (descending) for the 8 rows this lane sees (C layout).
  float t[8][KTOP];
#pragma unroll
  for (int r = 0; r < 8; ++r)
#pragma unroll
    for (int k = 0; k < KTOP; ++k) t[r][k] = -2.0f;

  issue_tile(cbeg, 0);
  wait_async0();
  __syncthreads();

  int cur = 0;
  for (int c0 = cbeg; c0 < cend; c0 += 16) {
    const int nxt = cur ^ 1;
    if (c0 + 16 < cend) issue_tile(c0 + 16, nxt);  // overlaps with WMMA below

    v8f acc = {0.f, 0.f, 0.f, 0.f, 0.f, 0.f, 0.f, 0.f};
#pragma unroll
    for (int c = 0; c < 16; ++c) {
      const int kb = half * 8 + c * 32;  // B 32x16 layout: lane holds K kb..kb+7 and kb+16..kb+23
      bfx8 h1 = *(const bfx8*)&Bt[cur][0][lan16][kb];
      bfx8 h2 = *(const bfx8*)&Bt[cur][0][lan16][kb + 16];
      bfx16 Bh = __builtin_shufflevector(h1, h2, 0,1,2,3,4,5,6,7,8,9,10,11,12,13,14,15);
      bfx8 l1 = *(const bfx8*)&Bt[cur][1][lan16][kb];
      bfx8 l2 = *(const bfx8*)&Bt[cur][1][lan16][kb + 16];
      bfx16 Bl = __builtin_shufflevector(l1, l2, 0,1,2,3,4,5,6,7,8,9,10,11,12,13,14,15);
      // bf16x3: hi*hi + hi*lo + lo*hi, fp32 accumulate (fp32-equivalent dots)
      acc = __builtin_amdgcn_wmma_f32_16x16x32_bf16(false, Ah[c], false, Bh, (short)0, acc, false, false);
      acc = __builtin_amdgcn_wmma_f32_16x16x32_bf16(false, Ah[c], false, Bl, (short)0, acc, false, false);
      acc = __builtin_amdgcn_wmma_f32_16x16x32_bf16(false, Al[c], false, Bh, (short)0, acc, false, false);
    }

    const bool diag = (c0 == m0);  // uniform branch condition
#pragma unroll
    for (int r = 0; r < 8; ++r) {
      // C layout: value r -> row m0 + r + 8*half, col c0 + lan16
      float vv = acc[r];
      vv = (diag && ((r + 8 * half) == lan16)) ? -1.0f : vv;
      float a = vv;  // branchless sorted insert (descending)
#pragma unroll
      for (int k = 0; k < KTOP; ++k) {
        float hi2 = fmaxf(t[r][k], a);
        a         = fminf(t[r][k], a);
        t[r][k]   = hi2;
      }
    }

    wait_async0();     // our copies for tile i+1 landed
    __syncthreads();   // everyone's copies landed; everyone done reading cur
    cur = nxt;
  }

  // ---- merge per-lane lists: row R's top-5 = top-5 over the 16 lanes' lists.
#pragma unroll
  for (int r = 0; r < 8; ++r)
#pragma unroll
    for (int k = 0; k < KTOP; ++k) sm[wave][r + 8 * half][lan16][k] = t[r][k];
  __syncthreads();

  if (lane < 16) {
    float b[KTOP];
#pragma unroll
    for (int k = 0; k < KTOP; ++k) b[k] = -2.0f;
    for (int j = 0; j < 16; ++j) {
#pragma unroll
      for (int k = 0; k < KTOP; ++k) {
        float a = sm[wave][lane][j][k];
#pragma unroll
        for (int q = 0; q < KTOP; ++q) {
          float hi2 = fmaxf(b[q], a);
          a         = fminf(b[q], a);
          b[q]      = hi2;
        }
      }
    }
    const int row = m0 + lane;
    float* dst = topvals + ((size_t)row * COLSPLITS + blockIdx.y) * KTOP;
#pragma unroll
    for (int k = 0; k < KTOP; ++k) dst[k] = b[k];
  }
}

// ---------------------------------------------------------------------------
// Kernel 3: per row, merge the 4 column-split top-5 lists (20 values),
// rho = mean_k sqrt(2 - 2*d_k), emit log(rho + eps).
// ---------------------------------------------------------------------------
__global__ __launch_bounds__(256) void rho_log_kernel(
    const float* __restrict__ topvals, float* __restrict__ logterm) {
  const int row = blockIdx.x * 256 + threadIdx.x;
  float b[KTOP];
#pragma unroll
  for (int k = 0; k < KTOP; ++k) b[k] = -2.0f;
  const float* src = topvals + (size_t)row * (COLSPLITS * KTOP);
#pragma unroll
  for (int j = 0; j < COLSPLITS * KTOP; ++j) {
    float a = src[j];
#pragma unroll
    for (int q = 0; q < KTOP; ++q) {
      float hi2 = fmaxf(b[q], a);
      a         = fminf(b[q], a);
      b[q]      = hi2;
    }
  }
  float s = 0.f;
#pragma unroll
  for (int k = 0; k < KTOP; ++k)
    s += sqrtf(fmaxf(2.0f - 2.0f * b[k], 0.0f));  // ||a-b|| on the unit sphere
  logterm[row] = logf(s * (1.0f / KTOP) + 1e-8f);
}

// ---------------------------------------------------------------------------
// Kernel 4: deterministic single-block reduction; loss = -mean(logterm).
// ---------------------------------------------------------------------------
__global__ __launch_bounds__(256) void reduce_kernel(
    const float* __restrict__ logterm, float* __restrict__ out) {
  __shared__ float sm[256];
  float s = 0.f;
  for (int i = threadIdx.x; i < N; i += 256) s += logterm[i];
  sm[threadIdx.x] = s;
  __syncthreads();
  for (int off = 128; off >= 1; off >>= 1) {
    if ((int)threadIdx.x < off) sm[threadIdx.x] += sm[threadIdx.x + off];
    __syncthreads();
  }
  if (threadIdx.x == 0) out[0] = -sm[0] / (float)N;
}

extern "C" void kernel_launch(void* const* d_in, const int* in_sizes, int n_in,
                              void* d_out, int out_size, void* d_ws, size_t ws_size,
                              hipStream_t stream) {
  const float* x = (const float*)d_in[0];
  char* ws = (char*)d_ws;
  __bf16* xh      = (__bf16*)ws;                                  // 8 MB
  __bf16* xl      = (__bf16*)(ws + (size_t)N * D * 2);            // 8 MB
  float*  topvals = (float*)(ws + (size_t)N * D * 4);             // 640 KB
  float*  logterm = (float*)(ws + (size_t)N * D * 4 +
                             (size_t)N * COLSPLITS * KTOP * 4);   // 32 KB

  normalize_split_kernel<<<N / 8, 256, 0, stream>>>(x, xh, xl);
  gram_topk_kernel<<<dim3(N / 128, COLSPLITS), 256, 0, stream>>>(xh, xl, topvals);
  rho_log_kernel<<<N / 256, 256, 0, stream>>>(topvals, logterm);
  reduce_kernel<<<1, 256, 0, stream>>>(logterm, (float*)d_out);
}
